// GAT_LSTM_29557964931465
// MI455X (gfx1250) — matrix operand order repaired and checked by reading, not compiled, not run on hardware
//
#include <hip/hip_runtime.h>
#include <hip/hip_bf16.h>

// ---------------- problem constants ----------------
#define Bb   4
#define Tt   256
#define Nn   48
#define Ff   4
#define Hh   64
#define Kk   4
#define Ee   192
#define KH   256      // K*H
#define LHd  512
#define G4   2048     // 4*LH
#define NH   3072     // N*H (LSTM input width)
#define Cc   8
#define NEG_SLOPE 0.2f

typedef __bf16 bf16_t;
typedef __attribute__((ext_vector_type(16))) __bf16 v16bf;
typedef __attribute__((ext_vector_type(8)))  float  v8f;

// ---------------- WMMA helper (CDNA5 16x16x32 bf16, f32 acc) ----------------
__device__ __forceinline__ v8f wmma_bf16(v16bf a, v16bf b, v8f c) {
  return __builtin_amdgcn_wmma_f32_16x16x32_bf16(false, a, false, b, (short)0, c,
                                                 false, false);
}

// Packed fragment layouts (ISA 7.12.2, bf16):
//  A tile 16x32 at (mt,kt): lane = hi*16 + (m&15); element e -> k-offset
//     koff = (j<4 ? 2j : 16+2(j-4)) + 8*hi + w   (j=e>>1, w=e&1)
//  B tile 32x16 at (kt,nt): lane = hi*16 + (n&15); e -> same koff(e,hi) as A
// Both stored so one lane's 16 bf16 live at a contiguous, 32B-aligned address.

// inverse map: (m,k) -> flat packed-A index (KT = K/32 tiles along K)
__device__ __forceinline__ size_t a_pack_idx(int m, int k, int K) {
  const int mt = m >> 4, lanelo = m & 15;
  const int kt = k >> 5, koff = k & 31;
  const int hi = (koff >> 3) & 1;            // koff 8-15 / 24-31 -> upper half-wave
  const int base = koff - (hi << 3);         // {0..7, 16..23}
  const int j = (base < 8) ? (base >> 1) : (4 + ((base - 16) >> 1));
  const int e = (j << 1) | (base & 1);
  const int lane = (hi << 4) | lanelo;
  return (((size_t)mt * (K >> 5) + kt) * 32 + lane) * 16 + e;
}

__device__ __forceinline__ v16bf frag_ld(const bf16_t* p) {
  return *(const v16bf*)p;                   // 32B: two b128 loads
}

__device__ __forceinline__ float sigm(float x) { return 1.0f / (1.0f + __expf(-x)); }
__device__ __forceinline__ float lrelu(float x) { return x > 0.0f ? x : NEG_SLOPE * x; }

// ---------------- weight packing: f32 -> packed-B bf16 fragments ----------------
// Logical B[K,N]; B[k][n] = transposed ? src[n*K + k] : src[k*N + n]
__global__ void k_pack_b(const float* __restrict__ src, bf16_t* __restrict__ dst,
                         int K, int N, int transposed) {
  const int i = blockIdx.x * 256 + threadIdx.x;
  if (i >= K * N) return;
  const int e = i & 15, lane = (i >> 4) & 31, tile = i >> 9;
  const int NT = N >> 4;
  const int nt = tile % NT, kt = tile / NT;
  const int k = kt * 32 + ((e >> 3) << 4) + ((lane >> 4) << 3) + (e & 7);
  const int n = nt * 16 + (lane & 15);
  const float v = transposed ? src[(size_t)n * K + k] : src[(size_t)k * N + n];
  dst[i] = (bf16_t)v;
}

// ---------------- GATv2 layer 1 (F=4 -> 256, concat) : one block per graph ----------------
// Output written directly in packed-A fragment order (rows = g*48+n, K = 256).
__global__ void k_gat1(const float* __restrict__ x, const int* __restrict__ ei,
                       const float* __restrict__ Wl, const float* __restrict__ Wr,
                       const float* __restrict__ att, const float* __restrict__ bias,
                       bf16_t* __restrict__ h1p) {
  __shared__ float s_x[Nn * Ff];
  __shared__ float s_wl[Ff * KH];
  __shared__ float s_att[KH];
  __shared__ float s_xl[Nn * KH];     // 48 KB
  __shared__ float s_e[Ee * Kk];
  __shared__ float s_max[Nn * Kk];
  __shared__ float s_den[Nn * Kk];
  __shared__ int   s_src[Ee], s_dst[Ee];

  const int g = blockIdx.x, tid = threadIdx.x;
  for (int i = tid; i < Nn * Ff; i += 256) s_x[i] = x[(size_t)g * Nn * Ff + i];
  for (int i = tid; i < Ff * KH; i += 256) s_wl[i] = Wl[i];
  for (int i = tid; i < KH; i += 256) s_att[i] = att[i];
  for (int i = tid; i < Ee; i += 256) { s_src[i] = ei[i]; s_dst[i] = ei[Ee + i]; }
  __syncthreads();

  for (int i = tid; i < Nn * KH; i += 256) {       // xl = x @ Wl
    const int n = i >> 8, c = i & 255;
    float s = 0.0f;
#pragma unroll
    for (int f = 0; f < Ff; ++f) s += s_x[n * Ff + f] * s_wl[f * KH + c];
    s_xl[i] = s;
  }
  __syncthreads();

  for (int i = tid; i < Ee * Kk; i += 256) {       // edge scores
    const int e = i >> 2, k = i & 3;
    const int sn = s_src[e], dn = s_dst[e];
    float acc = 0.0f;
    for (int h = 0; h < Hh; ++h) {
      const int c = k * Hh + h;
      float xr = 0.0f;
#pragma unroll
      for (int f = 0; f < Ff; ++f) xr += s_x[dn * Ff + f] * Wr[f * KH + c];
      acc += lrelu(s_xl[sn * KH + c] + xr) * s_att[c];
    }
    s_e[i] = acc;
  }
  __syncthreads();

  for (int i = tid; i < Nn * Kk; i += 256) {       // segment max
    const int n = i >> 2, k = i & 3;
    float m = -3.0e38f;
    for (int e = 0; e < Ee; ++e)
      if (s_dst[e] == n) { float v = s_e[e * Kk + k]; m = v > m ? v : m; }
    s_max[i] = m;
  }
  __syncthreads();
  for (int i = tid; i < Ee * Kk; i += 256) {
    const int e = i >> 2, k = i & 3;
    s_e[i] = __expf(s_e[i] - s_max[s_dst[e] * Kk + k]);
  }
  __syncthreads();
  for (int i = tid; i < Nn * Kk; i += 256) {       // denom
    const int n = i >> 2, k = i & 3;
    float s = 0.0f;
    for (int e = 0; e < Ee; ++e)
      if (s_dst[e] == n) s += s_e[e * Kk + k];
    s_den[i] = s + 1e-16f;
  }
  __syncthreads();
  for (int i = tid; i < Ee * Kk; i += 256) {       // alpha
    const int e = i >> 2, k = i & 3;
    s_e[i] = s_e[i] / s_den[s_dst[e] * Kk + k];
  }
  __syncthreads();

  for (int i = tid; i < Nn * KH; i += 256) {       // aggregate, +bias, relu
    const int n = i >> 8, c = i & 255, k = c >> 6;
    float s = 0.0f;
    for (int e = 0; e < Ee; ++e)
      if (s_dst[e] == n) s += s_e[e * Kk + k] * s_xl[s_src[e] * KH + c];
    s += bias[c];
    h1p[a_pack_idx(g * Nn + n, c, KH)] = (bf16_t)(s > 0.0f ? s : 0.0f);
  }
}

// ---------------- GATv2 layer 2 (256 -> 64, mean heads) : one block / graph, WMMA --------
__global__ void k_gat2(const bf16_t* __restrict__ h1p, const int* __restrict__ ei,
                       const bf16_t* __restrict__ Wl2p, const bf16_t* __restrict__ Wr2p,
                       const float* __restrict__ att, const float* __restrict__ bias,
                       bf16_t* __restrict__ seqp) {
  __shared__ bf16_t s_xl[Nn * KH];    // 24 KB (row-major f.p. results)
  __shared__ bf16_t s_xr[Nn * KH];    // 24 KB
  __shared__ float  s_e[Ee * Kk];
  __shared__ float  s_max[Nn * Kk];
  __shared__ float  s_den[Nn * Kk];
  __shared__ float  s_att[KH];
  __shared__ int    s_src[Ee], s_dst[Ee];

  const int g = blockIdx.x, tid = threadIdx.x;
  const int lane = tid & 31, wave = tid >> 5;   // 8 waves

  for (int i = tid; i < KH; i += 256) s_att[i] = att[i];
  for (int i = tid; i < Ee; i += 256) { s_src[i] = ei[i]; s_dst[i] = ei[Ee + i]; }

  // xl2/xr2 = h1 @ {Wl2,Wr2}: M=48 (3 tiles), N=256 (16 tiles), K=256 (8 WMMA)
  for (int task = wave; task < 96; task += 8) {
    const int mat = task & 1;
    const int t2 = task >> 1;
    const int mt = t2 >> 4, nt = t2 & 15;
    const bf16_t* Bp = mat ? Wr2p : Wl2p;
    v8f acc = {};
#pragma unroll
    for (int kt = 0; kt < 8; ++kt) {
      v16bf a = frag_ld(h1p + ((((size_t)g * 3 + mt) * 8 + kt) * 32 + lane) * 16);
      v16bf b = frag_ld(Bp + (((size_t)kt * 16 + nt) * 32 + lane) * 16);
      acc = wmma_bf16(a, b, acc);
    }
    bf16_t* dstS = mat ? s_xr : s_xl;
    const int n = nt * 16 + (lane & 15);
    const int mb = mt * 16 + ((lane >> 4) << 3);
#pragma unroll
    for (int r = 0; r < 8; ++r) dstS[(mb + r) * KH + n] = (bf16_t)acc[r];
  }
  __syncthreads();

  for (int i = tid; i < Ee * Kk; i += 256) {       // attention scores
    const int e = i >> 2, k = i & 3;
    const int sn = s_src[e], dn = s_dst[e];
    float acc = 0.0f;
    for (int h = 0; h < Hh; ++h) {
      const int c = k * Hh + h;
      acc += lrelu((float)s_xl[sn * KH + c] + (float)s_xr[dn * KH + c]) * s_att[c];
    }
    s_e[i] = acc;
  }
  __syncthreads();
  for (int i = tid; i < Nn * Kk; i += 256) {
    const int n = i >> 2, k = i & 3;
    float m = -3.0e38f;
    for (int e = 0; e < Ee; ++e)
      if (s_dst[e] == n) { float v = s_e[e * Kk + k]; m = v > m ? v : m; }
    s_max[i] = m;
  }
  __syncthreads();
  for (int i = tid; i < Ee * Kk; i += 256) {
    const int e = i >> 2, k = i & 3;
    s_e[i] = __expf(s_e[i] - s_max[s_dst[e] * Kk + k]);
  }
  __syncthreads();
  for (int i = tid; i < Nn * Kk; i += 256) {
    const int n = i >> 2, k = i & 3;
    float s = 0.0f;
    for (int e = 0; e < Ee; ++e)
      if (s_dst[e] == n) s += s_e[e * Kk + k];
    s_den[i] = s + 1e-16f;
  }
  __syncthreads();
  for (int i = tid; i < Ee * Kk; i += 256) {
    const int e = i >> 2, k = i & 3;
    s_e[i] = s_e[i] / s_den[s_dst[e] * Kk + k];
  }
  __syncthreads();

  // aggregate, mean over heads, +bias, relu -> seq (packed-A rows g, K=3072)
  for (int i = tid; i < Nn * Hh; i += 256) {
    const int n = i >> 6, h = i & 63;
    float s = 0.0f;
#pragma unroll
    for (int k = 0; k < Kk; ++k) {
      const int c = k * Hh + h;
      float sk = 0.0f;
      for (int e = 0; e < Ee; ++e)
        if (s_dst[e] == n) sk += s_e[e * Kk + k] * (float)s_xl[s_src[e] * KH + c];
      s += sk;
    }
    s = s * 0.25f + bias[h];
    seqp[a_pack_idx(g, i, NH)] = (bf16_t)(s > 0.0f ? s : 0.0f);
  }
}

// ---------------- LSTM input GEMM: XW[1024,2048] = seq[1024,3072] @ W_ih^T + bias -------
__global__ void k_lstm_in(const bf16_t* __restrict__ seqp, const bf16_t* __restrict__ Wp,
                          const float* __restrict__ b_ih, const float* __restrict__ b_hh,
                          float* __restrict__ xw) {
  const int lane = threadIdx.x & 31, wave = threadIdx.x >> 5;
  const int mt = blockIdx.x;                 // 64 M-tiles
  const int nt = blockIdx.y * 8 + wave;      // 128 N-tiles
  const int KT = NH / 32;                    // 96
  const bf16_t* ap = seqp + (((size_t)mt * KT) * 32 + lane) * 16;
  v8f acc = {};
  for (int kt = 0; kt < KT; ++kt) {
    const bf16_t* bp = Wp + (((size_t)kt * 128 + nt) * 32 + lane) * 16;
    if (kt + 1 < KT)
      __builtin_prefetch(Wp + (((size_t)(kt + 1) * 128 + nt) * 32 + lane) * 16, 0, 1);
    v16bf a = frag_ld(ap + (size_t)kt * 32 * 16);
    v16bf b = frag_ld(bp);
    acc = wmma_bf16(a, b, acc);
  }
  const int n = nt * 16 + (lane & 15);
  const int mb = mt * 16 + ((lane >> 4) << 3);
  const float bsum = b_ih[n] + b_hh[n];
#pragma unroll
  for (int r = 0; r < 8; ++r)
    xw[(size_t)(mb + r) * G4 + n] = acc[r] + bsum;
}

// ---------------- persistent LSTM scan (single block, 8 waves) + final FC ----------------
__global__ void k_lstm_scan(const float* __restrict__ xw, const bf16_t* __restrict__ Whhp,
                            const float* __restrict__ fc_w, const float* __restrict__ fc_b,
                            float* __restrict__ out) {
  __shared__ float  s_h[Bb * LHd];            // 8 KB
  __shared__ float  s_c[Bb * LHd];            // 8 KB
  __shared__ bf16_t s_hb[16 * LHd];           // 16 KB packed-A h (rows 4..15 zero)
  __shared__ float  s_g[Bb * G4];             // 32 KB gate buffer

  const int tid = threadIdx.x, lane = tid & 31, wave = tid >> 5;
  for (int i = tid; i < Bb * LHd; i += 256) { s_h[i] = 0.0f; s_c[i] = 0.0f; }
  for (int i = tid; i < 16 * LHd; i += 256) s_hb[i] = (bf16_t)0.0f;
  __syncthreads();

  for (int t = 0; t < Tt; ++t) {
    // G[pad16,2048] = h_pad @ W_hh^T ; 128 N-tiles, 16 per wave, K=512 (16 WMMA)
    for (int nt = wave; nt < G4 / 16; nt += 8) {
      v8f acc = {};
#pragma unroll
      for (int kt = 0; kt < LHd / 32; ++kt) {
        v16bf a = frag_ld(s_hb + ((kt * 32 + lane) << 4));             // ds_load_b128 x2
        v16bf b = frag_ld(Whhp + (((size_t)kt * 128 + nt) * 32 + lane) * 16);
        acc = wmma_bf16(a, b, acc);
      }
      const int n = nt * 16 + (lane & 15);
      if ((lane >> 4) == 0) {
#pragma unroll
        for (int r = 0; r < Bb; ++r)
          s_g[r * G4 + n] = acc[r] + xw[(size_t)(r * Tt + t) * G4 + n];
      }
    }
    __syncthreads();

    // gates: i,f,g,o split of 2048
    for (int i = tid; i < Bb * LHd; i += 256) {
      const int b = i >> 9, j = i & 511;
      const float ig = s_g[b * G4 + j];
      const float fg = s_g[b * G4 + LHd + j];
      const float gg = s_g[b * G4 + 2 * LHd + j];
      const float og = s_g[b * G4 + 3 * LHd + j];
      const float c = sigm(fg) * s_c[i] + sigm(ig) * tanhf(gg);
      const float h = sigm(og) * tanhf(c);
      s_c[i] = c; s_h[i] = h;
      s_hb[a_pack_idx(b, j, LHd)] = (bf16_t)h;     // packed-A store for next step
    }
    __syncthreads();
  }

  if (tid < Bb * Cc) {                             // out[4,8] = h @ fc_w^T + fc_b
    const int b = tid >> 3, c = tid & 7;
    float s = fc_b[c];
    for (int j = 0; j < LHd; ++j) s += s_h[b * LHd + j] * fc_w[c * LHd + j];
    out[b * Cc + c] = s;
  }
}

// ---------------- host entry ----------------
extern "C" void kernel_launch(void* const* d_in, const int* in_sizes, int n_in,
                              void* d_out, int out_size, void* d_ws, size_t ws_size,
                              hipStream_t stream) {
  (void)in_sizes; (void)n_in; (void)out_size; (void)ws_size;
  const float* x    = (const float*)d_in[0];
  const int*   ei   = (const int*)d_in[1];
  const float* Wl1  = (const float*)d_in[2];
  const float* Wr1  = (const float*)d_in[3];
  const float* att1 = (const float*)d_in[4];
  const float* b1   = (const float*)d_in[5];
  const float* Wl2  = (const float*)d_in[6];
  const float* Wr2  = (const float*)d_in[7];
  const float* att2 = (const float*)d_in[8];
  const float* b2   = (const float*)d_in[9];
  const float* W_ih = (const float*)d_in[10];
  const float* W_hh = (const float*)d_in[11];
  const float* b_ih = (const float*)d_in[12];
  const float* b_hh = (const float*)d_in[13];
  const float* fc_w = (const float*)d_in[14];
  const float* fc_b = (const float*)d_in[15];
  float* out = (float*)d_out;

  char* ws = (char*)d_ws;
  size_t off = 0;
  auto take = [&](size_t bytes) -> char* {
    char* p = ws + off;
    off += (bytes + 255) & ~(size_t)255;
    return p;
  };
  bf16_t* wl2p = (bf16_t*)take((size_t)KH * KH * 2);
  bf16_t* wr2p = (bf16_t*)take((size_t)KH * KH * 2);
  bf16_t* wihp = (bf16_t*)take((size_t)NH * G4 * 2);
  bf16_t* whhp = (bf16_t*)take((size_t)LHd * G4 * 2);
  bf16_t* h1p  = (bf16_t*)take((size_t)Bb * Tt * Nn * KH * 2);
  bf16_t* seqp = (bf16_t*)take((size_t)Bb * Tt * NH * 2);
  float*  xwb  = (float*)take((size_t)Bb * Tt * G4 * 4);

  const int G = Bb * Tt;

  // weight packing into WMMA B-fragment order (bf16)
  k_pack_b<<<(KH * KH + 255) / 256, 256, 0, stream>>>(Wl2, wl2p, KH, KH, 0);
  k_pack_b<<<(KH * KH + 255) / 256, 256, 0, stream>>>(Wr2, wr2p, KH, KH, 0);
  k_pack_b<<<(NH * G4 + 255) / 256, 256, 0, stream>>>(W_ih, wihp, NH, G4, 1);
  k_pack_b<<<(LHd * G4 + 255) / 256, 256, 0, stream>>>(W_hh, whhp, LHd, G4, 1);

  // GAT layers (layer-1 VALU, layer-2 WMMA)
  k_gat1<<<G, 256, 0, stream>>>(x, ei, Wl1, Wr1, att1, b1, h1p);
  k_gat2<<<G, 256, 0, stream>>>(h1p, ei, wl2p, wr2p, att2, b2, seqp);

  // LSTM input projection (WMMA GEMM) + persistent scan + FC
  k_lstm_in<<<dim3(G / 16, 16), 256, 0, stream>>>(seqp, wihp, b_ih, b_hh, xwb);
  k_lstm_scan<<<1, 256, 0, stream>>>(xwb, whhp, fc_w, fc_b, out);
}
